// MultiScaleRetention_37434934952284
// MI455X (gfx1250) — matrix operand order repaired and checked by reading, not compiled, not run on hardware
//
#include <hip/hip_runtime.h>
#include <hip/hip_bf16.h>

typedef __bf16 bf16;
typedef __attribute__((ext_vector_type(16))) __bf16 v16bf;
typedef __attribute__((ext_vector_type(8)))  __bf16 v8bf;
typedef __attribute__((ext_vector_type(8)))  float  v8f;

#define B_     4
#define L_     2048
#define SIZE_  512
#define H_     8
#define QK_    64
#define HS_    128
#define M_     (B_*L_)      /* 8192 */
#define SCALING_ 0.125f     /* 64^-0.5 */

// ---------------- WMMA helpers (CDNA5 gfx1250, wave32) ----------------

__device__ __forceinline__ v8f wmma_bf16(v16bf a, v16bf b, v8f c) {
  // v_wmma_f32_16x16x32_bf16: D = A(16x32) * B(32x16) + C(16x16 f32)
  return __builtin_amdgcn_wmma_f32_16x16x32_bf16(false, a, false, b, (short)0, c,
                                                 false, false);
}

// A fragment 16x32 bf16 from row-major memory (works for global and LDS).
// Layout: lanes 0-15 hold row m=lane with K = k0+0..15;
// lanes 16-31 hold row m=lane-16 with K = k0+16..31 (matches load_b_frag).
__device__ __forceinline__ v16bf load_a_frag(const bf16* base, int lda,
                                             int row0, int k0, int lane) {
  const int m  = lane & 15;
  const int kh = (lane >> 4) << 4; // 0 or 16
  const bf16* p = base + (size_t)(row0 + m) * lda + k0 + kh;
  union { v16bf v; v8bf h[2]; } u;
  u.h[0] = *(const v8bf*)(p);
  u.h[1] = *(const v8bf*)(p + 8);
  return u.v;
}

// B fragment 32x16 bf16 where memory is "NT": row n holds contiguous K.
// Layout (contiguous k, same as load_a_frag): lanes 0-15 -> column n=lane, K = k0+0..15;
//             lanes 16-31 -> column n=lane-16, K = k0+16..31.
__device__ __forceinline__ v16bf load_b_frag(const bf16* base, int ldb,
                                             int n0, int k0, int lane) {
  const int n  = lane & 15;
  const int kh = (lane >> 4) << 4; // 0 or 16
  return *(const v16bf*)(base + (size_t)(n0 + n) * ldb + k0 + kh);
}

// ---------------- pass 0: fp32 -> bf16 conversion ----------------

__global__ void f32_to_bf16(const float* __restrict__ src, bf16* __restrict__ dst, int n) {
  int i = blockIdx.x * blockDim.x + threadIdx.x;
  if (i < n) dst[i] = (bf16)src[i];
}

// ---------------- pass 1: fused QKVG projection GEMM ----------------
// C[m, n] = x[m, :] . Wcat[n, :] + bias, m in [0,8192), n in [0,3072)
// epilogue: n<512 -> q (b,h,l,d) ; <1024 -> k*SCALING (b,h,l,d) ;
//           <2048 -> v transposed (b,h,d,l) ; else silu(g) (m, 1024)

__global__ void __launch_bounds__(128) gemm_qkvg(
    const bf16* __restrict__ xb, const bf16* __restrict__ wcat,
    const float* __restrict__ bq, const float* __restrict__ bk,
    const float* __restrict__ bv, const float* __restrict__ bg,
    bf16* __restrict__ qbuf, bf16* __restrict__ kbuf,
    bf16* __restrict__ vT, bf16* __restrict__ gbuf) {
  const int lane  = threadIdx.x & 31;
  const int wave  = threadIdx.x >> 5;
  const int m0    = blockIdx.y * 64 + wave * 16;
  const int n0    = blockIdx.x * 64;
  const int mbase = (lane >> 4) << 3;
  const int nloc  = lane & 15;

  v8f acc[4];
  #pragma unroll
  for (int c = 0; c < 4; ++c)
    #pragma unroll
    for (int e = 0; e < 8; ++e) acc[c][e] = 0.f;

  for (int k0 = 0; k0 < SIZE_; k0 += 32) {
    v16bf a = load_a_frag(xb, SIZE_, m0, k0, lane);
    #pragma unroll
    for (int c = 0; c < 4; ++c) {
      v16bf b = load_b_frag(wcat, SIZE_, n0 + c * 16, k0, lane);
      acc[c] = wmma_bf16(a, b, acc[c]);
    }
  }

  #pragma unroll
  for (int c = 0; c < 4; ++c) {
    const int n = n0 + c * 16 + nloc;
    #pragma unroll
    for (int v = 0; v < 8; ++v) {
      const int m  = m0 + mbase + v;
      const int bb = m >> 11;        // / L_
      const int l  = m & (L_ - 1);
      float val = acc[c][v];
      if (n < 512) {
        val += bq[n];
        const int h = n >> 6, d = n & 63;
        qbuf[(((size_t)(bb * H_ + h)) * L_ + l) * QK_ + d] = (bf16)val;
      } else if (n < 1024) {
        const int nn = n - 512;
        val = (val + bk[nn]) * SCALING_;
        const int h = nn >> 6, d = nn & 63;
        kbuf[(((size_t)(bb * H_ + h)) * L_ + l) * QK_ + d] = (bf16)val;
      } else if (n < 2048) {
        const int nn = n - 1024;
        val += bv[nn];
        const int h = nn >> 7, d = nn & 127;
        vT[(((size_t)(bb * H_ + h)) * HS_ + d) * L_ + l] = (bf16)val; // transposed
      } else {
        const int nn = n - 2048;
        val += bg[nn];
        const float s = val / (1.f + __expf(-val)); // SiLU
        gbuf[(size_t)m * 1024 + nn] = (bf16)s;
      }
    }
  }
}

// ---------------- pass 2: in-place RoPE on q and k ----------------

__global__ void rope_inplace(bf16* __restrict__ qbuf, bf16* __restrict__ kbuf,
                             const float* __restrict__ cosT,
                             const float* __restrict__ sinT) {
  bf16* buf = blockIdx.y ? kbuf : qbuf;
  const int idx = blockIdx.x * blockDim.x + threadIdx.x; // pair index
  if (idx >= B_ * H_ * L_ * (QK_ / 2)) return;
  const int d2   = idx & 31;        // pair within head dim
  const int rest = idx >> 5;        // bh*L + l
  const int l    = rest & (L_ - 1);
  const size_t base = (size_t)rest * QK_ + d2 * 2;
  const float c = cosT[l * QK_ + d2 * 2]; // cos repeated in pairs
  const float s = sinT[l * QK_ + d2 * 2];
  const float e = (float)buf[base], o = (float)buf[base + 1];
  buf[base]     = (bf16)(e * c - o * s);
  buf[base + 1] = (bf16)(o * c + e * s);
}

// ---------------- pass 3: fused causal retention + LN + gate ----------------
// grid: (L/64 query tiles, B*H). 4 waves/block; wave owns a 16-row strip.

__global__ void __launch_bounds__(128) retention_attn(
    const bf16* __restrict__ qrot, const bf16* __restrict__ krot,
    const bf16* __restrict__ vT, const bf16* __restrict__ gsilu,
    bf16* __restrict__ obuf) {
  __shared__ bf16 lds_sm[4 * 16 * 64]; // per-wave masked-score tiles (bf16)

  const int lane  = threadIdx.x & 31;
  const int wave  = threadIdx.x >> 5;
  const int bh    = blockIdx.y;
  const int bb    = bh >> 3, h = bh & 7;
  const int i0    = blockIdx.x * 64;
  const int mbase = (lane >> 4) << 3;
  const int nloc  = lane & 15;

  const bf16* Q = qrot + (size_t)bh * L_ * QK_;
  const bf16* K = krot + (size_t)bh * L_ * QK_;
  const bf16* V = vT   + (size_t)bh * HS_ * L_;
  bf16* myS = lds_sm + wave * 16 * 64;

  // gamma_h = 1 - 2^{-5-h}; decay mask recomputed analytically (saves mask BW)
  const float one_mg = exp2f(-5.f - (float)h);      // 1 - gamma (exact)
  const float lg     = log2f(1.f - one_mg);         // log2(gamma) < 0
  const float inv1mg = 1.f / one_mg;

  int irow[8];
  float rnorm[8];
  #pragma unroll
  for (int v = 0; v < 8; ++v) {
    irow[v] = i0 + wave * 16 + mbase + v;
    const float Si = (1.f - exp2f(lg * (float)(irow[v] + 1))) * inv1mg;
    rnorm[v] = rsqrtf(fmaxf(Si, 1e-6f)); // row norm of decay matrix
  }
  float colfac[4];
  #pragma unroll
  for (int c = 0; c < 4; ++c)
    colfac[c] = exp2f(-lg * (float)(c * 16 + nloc)); // gamma^{-(col offset)}

  const v16bf qf0 = load_a_frag(Q, QK_, i0 + wave * 16, 0, lane);
  const v16bf qf1 = load_a_frag(Q, QK_, i0 + wave * 16, 32, lane);

  v8f o_acc[8];
  #pragma unroll
  for (int t = 0; t < 8; ++t)
    #pragma unroll
    for (int e = 0; e < 8; ++e) o_acc[t][e] = 0.f;
  float rowsum[8];
  #pragma unroll
  for (int v = 0; v < 8; ++v) rowsum[v] = 0.f;

  for (int j0 = 0; j0 <= i0; j0 += 64) {   // causal key-block loop
    // S = Qr @ Krᵀ  (16x64 strip per wave)
    v8f s[4];
    #pragma unroll
    for (int c = 0; c < 4; ++c) {
      #pragma unroll
      for (int e = 0; e < 8; ++e) s[c][e] = 0.f;
      v16bf b0 = load_b_frag(K, QK_, j0 + c * 16, 0, lane);
      s[c] = wmma_bf16(qf0, b0, s[c]);
      v16bf b1 = load_b_frag(K, QK_, j0 + c * 16, 32, lane);
      s[c] = wmma_bf16(qf1, b1, s[c]);
    }
    // decay mask + running row sum + stage to LDS as bf16 A-matrix data
    #pragma unroll
    for (int v = 0; v < 8; ++v) {
      const int   d0 = irow[v] - j0;                       // >= 0
      const float av = exp2f(lg * (float)d0) * rnorm[v];   // gamma^{i-j0} * rnorm_i
      #pragma unroll
      for (int c = 0; c < 4; ++c) {
        const int off = c * 16 + nloc;                     // j - j0
        const float sv = (off <= d0) ? s[c][v] * av * colfac[c] : 0.f;
        rowsum[v] += sv;
        myS[(mbase + v) * 64 + off] = (bf16)sv;
      }
    }
    __syncthreads(); // LDS store -> load boundary (uniform: bound dep on blockIdx only)

    // O += Sm @ V  (V pre-transposed to (d, l) so B-frags are contiguous)
    const v16bf a0 = load_a_frag(myS, 64, 0, 0, lane);
    const v16bf a1 = load_a_frag(myS, 64, 0, 32, lane);
    #pragma unroll
    for (int t = 0; t < 8; ++t) {
      v16bf vb0 = load_b_frag(V, L_, t * 16, j0, lane);
      o_acc[t] = wmma_bf16(a0, vb0, o_acc[t]);
      v16bf vb1 = load_b_frag(V, L_, t * 16, j0 + 32, lane);
      o_acc[t] = wmma_bf16(a1, vb1, o_acc[t]);
    }
    __syncthreads();
  }

  // denom = max(|row sum|, 1) ; reduce across the 16 n-lanes
  float denom[8];
  #pragma unroll
  for (int v = 0; v < 8; ++v) {
    float r = rowsum[v];
    r += __shfl_xor(r, 1); r += __shfl_xor(r, 2);
    r += __shfl_xor(r, 4); r += __shfl_xor(r, 8);
    denom[v] = 1.f / fmaxf(fabsf(r), 1.f);
  }

  // per-row LayerNorm over HS=128, then SiLU-gate and store bf16
  #pragma unroll
  for (int v = 0; v < 8; ++v) {
    float s1 = 0.f, s2 = 0.f;
    #pragma unroll
    for (int t = 0; t < 8; ++t) {
      const float x = o_acc[t][v] * denom[v];
      o_acc[t][v] = x;
      s1 += x; s2 += x * x;
    }
    s1 += __shfl_xor(s1, 1); s1 += __shfl_xor(s1, 2);
    s1 += __shfl_xor(s1, 4); s1 += __shfl_xor(s1, 8);
    s2 += __shfl_xor(s2, 1); s2 += __shfl_xor(s2, 2);
    s2 += __shfl_xor(s2, 4); s2 += __shfl_xor(s2, 8);
    const float mu  = s1 * (1.f / 128.f);
    const float var = s2 * (1.f / 128.f) - mu * mu;
    const float rstd = rsqrtf(var + 1e-5f);

    const size_t rowoff = ((size_t)(bb * L_ + irow[v])) * 1024 + h * HS_;
    #pragma unroll
    for (int t = 0; t < 8; ++t) {
      const int d = t * 16 + nloc;
      const float xn = (o_acc[t][v] - mu) * rstd;
      const float g  = (float)gsilu[rowoff + d];
      obuf[rowoff + d] = (bf16)(xn * g);
    }
  }
}

// ---------------- pass 4: output projection GEMM ----------------

__global__ void __launch_bounds__(128) gemm_out(
    const bf16* __restrict__ ob, const bf16* __restrict__ wob,
    const float* __restrict__ bo, float* __restrict__ out) {
  const int lane  = threadIdx.x & 31;
  const int wave  = threadIdx.x >> 5;
  const int m0    = blockIdx.y * 64 + wave * 16;
  const int n0    = blockIdx.x * 64;
  const int mbase = (lane >> 4) << 3;
  const int nloc  = lane & 15;

  v8f acc[4];
  #pragma unroll
  for (int c = 0; c < 4; ++c)
    #pragma unroll
    for (int e = 0; e < 8; ++e) acc[c][e] = 0.f;

  for (int k0 = 0; k0 < 1024; k0 += 32) {
    v16bf a = load_a_frag(ob, 1024, m0, k0, lane);
    #pragma unroll
    for (int c = 0; c < 4; ++c) {
      v16bf b = load_b_frag(wob, 1024, n0 + c * 16, k0, lane);
      acc[c] = wmma_bf16(a, b, acc[c]);
    }
  }

  #pragma unroll
  for (int c = 0; c < 4; ++c) {
    const int n = n0 + c * 16 + nloc;
    const float bias = bo[n];
    #pragma unroll
    for (int v = 0; v < 8; ++v) {
      const int m = m0 + mbase + v;
      out[(size_t)m * SIZE_ + n] = acc[c][v] + bias;
    }
  }
}

// ---------------- launch ----------------

extern "C" void kernel_launch(void* const* d_in, const int* in_sizes, int n_in,
                              void* d_out, int out_size, void* d_ws, size_t ws_size,
                              hipStream_t stream) {
  (void)in_sizes; (void)n_in; (void)out_size; (void)ws_size;
  const float* x    = (const float*)d_in[0];
  const float* cosT = (const float*)d_in[1];
  const float* sinT = (const float*)d_in[2];
  // d_in[3] = mask: unused (decay mask recomputed analytically in-kernel)
  const float* Wq = (const float*)d_in[4];  const float* bq = (const float*)d_in[5];
  const float* Wk = (const float*)d_in[6];  const float* bk = (const float*)d_in[7];
  const float* Wv = (const float*)d_in[8];  const float* bv = (const float*)d_in[9];
  const float* Wg = (const float*)d_in[10]; const float* bg = (const float*)d_in[11];
  const float* Wo = (const float*)d_in[12]; const float* bo = (const float*)d_in[13];
  float* out = (float*)d_out;

  char* ws = (char*)d_ws;
  bf16* xb   = (bf16*)(ws + 0);          // 8192*512  bf16 :  8 MB
  bf16* wcat = (bf16*)(ws + 8388608);    // 3072*512  bf16 :  3 MB
  bf16* wob  = (bf16*)(ws + 11534336);   // 512*1024  bf16 :  1 MB
  bf16* qbuf = (bf16*)(ws + 12582912);   // (b,h,l,64)     :  8 MB
  bf16* kbuf = (bf16*)(ws + 20971520);   // (b,h,l,64)     :  8 MB
  bf16* vT   = (bf16*)(ws + 29360128);   // (b,h,128,l)    : 16 MB
  bf16* gbuf = (bf16*)(ws + 46137344);   // silu(g) (m,1024): 16 MB
  bf16* obuf = (bf16*)(ws + 62914560);   // gated o (m,1024): 16 MB

  f32_to_bf16<<<(M_*SIZE_ + 255) / 256, 256, 0, stream>>>(x, xb, M_*SIZE_);
  f32_to_bf16<<<(262144 + 255) / 256, 256, 0, stream>>>(Wq, wcat,           262144);
  f32_to_bf16<<<(262144 + 255) / 256, 256, 0, stream>>>(Wk, wcat + 262144,  262144);
  f32_to_bf16<<<(524288 + 255) / 256, 256, 0, stream>>>(Wv, wcat + 524288,  524288);
  f32_to_bf16<<<(524288 + 255) / 256, 256, 0, stream>>>(Wg, wcat + 1048576, 524288);
  f32_to_bf16<<<(524288 + 255) / 256, 256, 0, stream>>>(Wo, wob,            524288);

  gemm_qkvg<<<dim3(48, 128), 128, 0, stream>>>(xb, wcat, bq, bk, bv, bg,
                                               qbuf, kbuf, vT, gbuf);
  rope_inplace<<<dim3(8192, 2), 256, 0, stream>>>(qbuf, kbuf, cosT, sinT);
  retention_attn<<<dim3(32, 32), 128, 0, stream>>>(qbuf, kbuf, vT, gbuf, obuf);
  gemm_out<<<dim3(8, 128), 128, 0, stream>>>(obuf, wob, bo, out);
}
